// Encoder_32384053412295
// MI455X (gfx1250) — matrix operand order repaired and checked by reading, not compiled, not run on hardware
//
#include <hip/hip_runtime.h>
#include <hip/hip_bf16.h>

// ---------------------------------------------------------------------------
// Encoder self-attention block for MI455X (gfx1250), bf16 WMMA pipeline.
// B=4, S=2048, D=512, H=8, DH=64.  All matmuls on v_wmma_f32_16x16x32_bf16.
// Attention K/V tiles staged into LDS with the Tensor Data Mover (TDM).
// ---------------------------------------------------------------------------

constexpr int B_  = 4;
constexpr int S_  = 2048;
constexpr int D_  = 512;
constexpr int H_  = 8;
constexpr int DH_ = 64;

typedef __attribute__((ext_vector_type(16))) __bf16 v16bf;
typedef __attribute__((ext_vector_type(8)))  __bf16 bf16x8;
typedef __attribute__((ext_vector_type(8)))  float  v8f;

typedef unsigned int u32;
typedef u32 __attribute__((ext_vector_type(4))) u32x4;
typedef int __attribute__((ext_vector_type(4))) i32x4;
typedef int __attribute__((ext_vector_type(8))) i32x8;

__device__ __forceinline__ v16bf cat8(bf16x8 lo, bf16x8 hi) {
  return __builtin_shufflevector(lo, hi, 0, 1, 2, 3, 4, 5, 6, 7,
                                         8, 9, 10, 11, 12, 13, 14, 15);
}

// A-matrix (16x32, M x K) fragment: lane m = lane&15, half = lane>>4.
// VGPR pair p holds k = (p<4 ? 2p : 16+2(p-4)) + half*8  -> two contiguous
// 8-element runs: [half*8 .. half*8+7] and [16+half*8 .. 16+half*8+7].
__device__ __forceinline__ v16bf load_a_frag(const __bf16* row32, int hl) {
  bf16x8 lo = *(const bf16x8*)(row32 + hl * 8);
  bf16x8 hi = *(const bf16x8*)(row32 + 16 + hl * 8);
  return cat8(lo, hi);
}

// B-matrix (32x16, K x N) fragment: lane n = lane&15, half = lane>>4.
// Same k order as A: runs [half*8 .. half*8+7] and [16+half*8 .. 16+half*8+7]
// of the column, stored k-major for each column (we stage it that way).
__device__ __forceinline__ v16bf load_b_frag(const __bf16* col32, int hl) {
  bf16x8 lo = *(const bf16x8*)(col32 + hl * 8);
  bf16x8 hi = *(const bf16x8*)(col32 + 16 + hl * 8);
  return cat8(lo, hi);
}

__device__ __forceinline__ v8f wmma_bf16(v16bf a, v16bf b, v8f c) {
  return __builtin_amdgcn_wmma_f32_16x16x32_bf16(
      /*neg_a=*/false, a, /*neg_b=*/false, b,
      /*c_mod=*/(short)0, c, /*reuse_a=*/false, /*reuse_b=*/false);
}

// ---------------------------------------------------------------------------
// Tensor Data Mover: 2-D tile (bf16 elements) global -> LDS.
//   dim0 = contiguous dimension (elements), dim1 = rows.
//   D# group layout per CDNA5 ISA ch.8 (CDNA5 ISA).
// ---------------------------------------------------------------------------
#if __has_builtin(__builtin_amdgcn_tensor_load_to_lds)
#define HAVE_TDM 1
__device__ __forceinline__ void tdm_load_tile_2d_bf16(
    u32 lds_off, unsigned long long gaddr,
    u32 tile_d0, u32 tile_d1, u32 tensor_d0, u32 tensor_d1, u32 stride0) {
  u32x4 g0;
  g0.x = 1u;                                   // count=1, user mode
  g0.y = lds_off;                              // lds_addr (bytes)
  g0.z = (u32)(gaddr & 0xffffffffull);         // global_addr[31:0]
  g0.w = (u32)((gaddr >> 32) & 0x01ffffffull)  // global_addr[56:32]
         | (2u << 30);                         // type = 2 ("image")
  i32x8 g1;
  g1[0] = (int)(1u << 16);                     // data_size=1 -> 2 bytes
  g1[1] = (int)((tensor_d0 & 0xffffu) << 16);  // tensor_dim0[15:0]
  g1[2] = (int)(((tensor_d0 >> 16) & 0xffffu) |
                ((tensor_d1 & 0xffffu) << 16));
  g1[3] = (int)(((tensor_d1 >> 16) & 0xffffu) |
                ((tile_d0 & 0xffffu) << 16));  // tile_dim0
  g1[4] = (int)(tile_d1 & 0xffffu);            // tile_dim1 (tile_dim2 = 0)
  g1[5] = (int)stride0;                        // tensor_dim0_stride[31:0]
  g1[6] = 0;                                   // stride0 hi / dim1_stride lo
  g1[7] = 0;
  i32x4 z4 = {0, 0, 0, 0};
#if __clang_major__ >= 23
  i32x8 z8 = {0, 0, 0, 0, 0, 0, 0, 0};
  __builtin_amdgcn_tensor_load_to_lds(g0, g1, z4, z4, z8, 0);
#else
  __builtin_amdgcn_tensor_load_to_lds(g0, g1, z4, z4, 0);
#endif
}
#endif

// ---------------------------------------------------------------------------
// Casts
// ---------------------------------------------------------------------------
__global__ void cast_f32_bf16(const float* __restrict__ in,
                              __bf16* __restrict__ out, int n) {
  int i = blockIdx.x * blockDim.x + threadIdx.x;
  if (i < n) out[i] = (__bf16)in[i];
}

// in: [h][K][N] f32 -> out: [h][N][K] bf16 (so GEMM B operand is k-major)
__global__ void cast_transpose_bf16(const float* __restrict__ in,
                                    __bf16* __restrict__ out,
                                    int KN, int K, int N, int n) {
  int i = blockIdx.x * blockDim.x + threadIdx.x;
  if (i < n) {
    int h   = i / KN;
    int rem = i - h * KN;
    int k   = rem / N;
    int nn  = rem - k * N;
    out[(size_t)h * KN + (size_t)nn * K + k] = (__bf16)in[i];
  }
}

// ---------------------------------------------------------------------------
// QKV projections: per (b,h,sel) compute x[b] (SxD) @ W[h]^ (DxDH) -> SxDH.
// Block = 64 rows x 64 cols (DH), 4 waves, 16-row tile per wave.
// V is stored transposed ([dh][s]) for the attention PV stage.
// ---------------------------------------------------------------------------
__global__ __launch_bounds__(128) void qkv_gemm(
    const __bf16* __restrict__ xh,
    const __bf16* __restrict__ WqT, const __bf16* __restrict__ WkT,
    const __bf16* __restrict__ WvT,
    __bf16* __restrict__ Qo, __bf16* __restrict__ Ko,
    __bf16* __restrict__ VoT) {
  const int sel = blockIdx.z;
  const int bh  = blockIdx.y;
  const int b   = bh / H_;
  const int h   = bh % H_;
  const int tid = threadIdx.x;
  const int wv  = tid >> 5;
  const int ln  = tid & 31;
  const int hl  = ln >> 4;
  const int l15 = ln & 15;
  const int row0 = blockIdx.x * 64;

  const __bf16* W = (sel == 0 ? WqT : sel == 1 ? WkT : WvT) +
                    (size_t)h * D_ * DH_;            // [DH][D] k-major
  const __bf16* A = xh + (size_t)b * S_ * D_;        // [S][D] row-major

  __shared__ __bf16 As[64][32];
  __shared__ __bf16 Bt[64][32];

  v8f z = {};
  v8f acc[4];
#pragma unroll
  for (int t = 0; t < 4; ++t) acc[t] = z;

  const int ar = tid >> 1, aq = tid & 1;
  for (int kb = 0; kb < D_ / 32; ++kb) {
    __syncthreads();
    *(v16bf*)&As[ar][aq * 16] =
        *(const v16bf*)&A[(size_t)(row0 + ar) * D_ + kb * 32 + aq * 16];
    *(v16bf*)&Bt[ar][aq * 16] =
        *(const v16bf*)&W[(size_t)ar * D_ + kb * 32 + aq * 16];
    __syncthreads();

    v16bf aF = load_a_frag(&As[wv * 16 + l15][0], hl);
#pragma unroll
    for (int t = 0; t < 4; ++t)
      acc[t] = wmma_bf16(aF, load_b_frag(&Bt[t * 16 + l15][0], hl), acc[t]);
  }

  const size_t obase = (size_t)bh * S_ * DH_;
#pragma unroll
  for (int r = 0; r < 8; ++r) {
    const int row = row0 + wv * 16 + r + hl * 8;
#pragma unroll
    for (int t = 0; t < 4; ++t) {
      const int col = t * 16 + l15;
      const float v = acc[t][r];
      if (sel == 0)      Qo[obase + (size_t)row * DH_ + col] = (__bf16)v;
      else if (sel == 1) Ko[obase + (size_t)row * DH_ + col] = (__bf16)v;
      else               VoT[obase + (size_t)col * S_ + row] = (__bf16)v;
    }
  }
}

// ---------------------------------------------------------------------------
// Fused flash-attention: per (b,h), 64 queries per block (16 per wave),
// stream 32-key blocks with online softmax; WMMA for QK^T and PV.
// K/V tiles staged into LDS by the Tensor Data Mover (wave 0 issues,
// s_wait_tensorcnt, then workgroup barrier publishes the tiles).
// ---------------------------------------------------------------------------
__global__ __launch_bounds__(128) void attn_kernel(
    const __bf16* __restrict__ Q, const __bf16* __restrict__ K,
    const __bf16* __restrict__ VT, const int* __restrict__ mask,
    __bf16* __restrict__ ctx) {
  const int bh  = blockIdx.y;
  const int b   = bh / H_;
  const int h   = bh % H_;
  const int tid = threadIdx.x;
  const int wv  = tid >> 5;
  const int ln  = tid & 31;
  const int hl  = ln >> 4;
  const int l15 = ln & 15;

  __shared__ __bf16 Ks[32][64];       // [key][dh]  (k-major for QK^T B-frag)
  __shared__ __bf16 Vt[64][32];       // [dh][key]  (k-major for PV  B-frag)
  __shared__ __bf16 Pw[4][16][32];    // per-wave P scratch (C->A relayout)

  const __bf16* Qb = Q  + (size_t)bh * S_ * DH_;
  const __bf16* Kb = K  + (size_t)bh * S_ * DH_;
  const __bf16* Vb = VT + (size_t)bh * S_ * DH_;   // [DH][S]
  const int* mk = mask + b * S_;

  const int q0 = blockIdx.x * 64 + wv * 16;
  const int am = q0 + l15;

  const v16bf aQ0 = load_a_frag(&Qb[(size_t)am * DH_ + 0], hl);
  const v16bf aQ1 = load_a_frag(&Qb[(size_t)am * DH_ + 32], hl);

  v8f z = {};
  v8f accO[4];
#pragma unroll
  for (int t = 0; t < 4; ++t) accO[t] = z;
  float mx[8], sm[8];
#pragma unroll
  for (int r = 0; r < 8; ++r) { mx[r] = -1e30f; sm[r] = 0.0f; }

  const float scale = 0.125f;  // 1/sqrt(DH)

#if defined(HAVE_TDM)
  const u32 ks_lds = (u32)(unsigned long long)(uintptr_t)&Ks[0][0];
  const u32 vt_lds = (u32)(unsigned long long)(uintptr_t)&Vt[0][0];
#else
  const int kr = tid >> 2, kq = tid & 3;   // Ks staging indices
  const int dr = tid >> 1, dq = tid & 1;   // Vt staging indices
#endif

  for (int kb = 0; kb < S_ / 32; ++kb) {
    __syncthreads();
#if defined(HAVE_TDM)
    if (wv == 0) {
      // K tile: 32 key-rows x 64 dh (dh contiguous) from [S][DH] tensor
      tdm_load_tile_2d_bf16(
          ks_lds,
          (unsigned long long)(uintptr_t)(Kb + (size_t)kb * 32 * DH_),
          /*tile_d0=*/DH_, /*tile_d1=*/32,
          /*tensor_d0=*/DH_, /*tensor_d1=*/S_, /*stride0=*/DH_);
      // V tile: 64 dh-rows x 32 keys (keys contiguous) from [DH][S] tensor
      tdm_load_tile_2d_bf16(
          vt_lds,
          (unsigned long long)(uintptr_t)(Vb + (size_t)kb * 32),
          /*tile_d0=*/32, /*tile_d1=*/DH_,
          /*tensor_d0=*/S_, /*tensor_d1=*/DH_, /*stride0=*/S_);
      __builtin_amdgcn_s_wait_tensorcnt(0);
    }
#else
    *(v16bf*)&Ks[kr][kq * 16] =
        *(const v16bf*)&Kb[(size_t)(kb * 32 + kr) * DH_ + kq * 16];
    *(v16bf*)&Vt[dr][dq * 16] =
        *(const v16bf*)&Vb[(size_t)dr * S_ + kb * 32 + dq * 16];
    if (kb + 1 < S_ / 32) {
      __builtin_prefetch(&Kb[(size_t)(kb * 32 + 32 + kr) * DH_], 0, 0);
      __builtin_prefetch(&Vb[(size_t)dr * S_ + kb * 32 + 32], 0, 0);
    }
#endif
    __syncthreads();

    // scores: two 16x16 key tiles, K-dim = DH = 64 (two 32-steps)
    v8f s0 = z, s1 = z;
    s0 = wmma_bf16(aQ0, load_b_frag(&Ks[l15][0],       hl), s0);
    s0 = wmma_bf16(aQ1, load_b_frag(&Ks[l15][32],      hl), s0);
    s1 = wmma_bf16(aQ0, load_b_frag(&Ks[16 + l15][0],  hl), s1);
    s1 = wmma_bf16(aQ1, load_b_frag(&Ks[16 + l15][32], hl), s1);

    const int key0 = kb * 32 + l15;
    const float bias0 = (mk[key0]      > 0) ? 0.0f : -1.0e9f;
    const float bias1 = (mk[key0 + 16] > 0) ? 0.0f : -1.0e9f;

    float p0[8], p1[8];
#pragma unroll
    for (int r = 0; r < 8; ++r) {
      float v0 = s0[r] * scale + bias0;
      float v1 = s1[r] * scale + bias1;
      float vm = fmaxf(v0, v1);
#pragma unroll
      for (int off = 8; off >= 1; off >>= 1)
        vm = fmaxf(vm, __shfl_xor(vm, off, 16));
      float mnew = fmaxf(mx[r], vm);
      float fac  = __expf(mx[r] - mnew);
      mx[r] = mnew;
      p0[r] = __expf(v0 - mnew);
      p1[r] = __expf(v1 - mnew);
      float rs = p0[r] + p1[r];
#pragma unroll
      for (int off = 8; off >= 1; off >>= 1)
        rs += __shfl_xor(rs, off, 16);
      sm[r] = sm[r] * fac + rs;
#pragma unroll
      for (int t = 0; t < 4; ++t) accO[t][r] *= fac;
    }

    // C-layout (m = r + 8*hl, n = l15) -> A-layout via per-wave LDS scratch
#pragma unroll
    for (int r = 0; r < 8; ++r) {
      Pw[wv][r + hl * 8][l15]      = (__bf16)p0[r];
      Pw[wv][r + hl * 8][16 + l15] = (__bf16)p1[r];
    }
    const v16bf aP = load_a_frag(&Pw[wv][l15][0], hl);

    // O += P (16x32) @ V (32x64): 4 n-tiles
#pragma unroll
    for (int t = 0; t < 4; ++t)
      accO[t] = wmma_bf16(aP, load_b_frag(&Vt[t * 16 + l15][0], hl), accO[t]);
  }

#pragma unroll
  for (int r = 0; r < 8; ++r) {
    const float inv = 1.0f / sm[r];
    const int row = q0 + r + hl * 8;
    const size_t base = ((size_t)b * S_ + row) * D_ + h * DH_;
#pragma unroll
    for (int t = 0; t < 4; ++t)
      ctx[base + t * 16 + l15] = (__bf16)(accO[t][r] * inv);
  }
}

// ---------------------------------------------------------------------------
// Output projection: ctx (B*S x 512, bf16) @ Wo^T-staged -> out (f32)
// ---------------------------------------------------------------------------
__global__ __launch_bounds__(128) void out_gemm(
    const __bf16* __restrict__ ctx, const __bf16* __restrict__ WoT,
    float* __restrict__ out) {
  const int tid = threadIdx.x;
  const int wv  = tid >> 5;
  const int ln  = tid & 31;
  const int hl  = ln >> 4;
  const int l15 = ln & 15;
  const int row0 = blockIdx.x * 64;
  const int col0 = blockIdx.y * 64;

  __shared__ __bf16 As[64][32];
  __shared__ __bf16 Bt[64][32];

  v8f z = {};
  v8f acc[4];
#pragma unroll
  for (int t = 0; t < 4; ++t) acc[t] = z;

  const int ar = tid >> 1, aq = tid & 1;
  for (int kb = 0; kb < D_ / 32; ++kb) {
    __syncthreads();
    *(v16bf*)&As[ar][aq * 16] =
        *(const v16bf*)&ctx[(size_t)(row0 + ar) * D_ + kb * 32 + aq * 16];
    *(v16bf*)&Bt[ar][aq * 16] =
        *(const v16bf*)&WoT[(size_t)(col0 + ar) * D_ + kb * 32 + aq * 16];
    __syncthreads();

    v16bf aF = load_a_frag(&As[wv * 16 + l15][0], hl);
#pragma unroll
    for (int t = 0; t < 4; ++t)
      acc[t] = wmma_bf16(aF, load_b_frag(&Bt[t * 16 + l15][0], hl), acc[t]);
  }

#pragma unroll
  for (int r = 0; r < 8; ++r) {
    const int row = row0 + wv * 16 + r + hl * 8;
#pragma unroll
    for (int t = 0; t < 4; ++t)
      out[(size_t)row * D_ + col0 + t * 16 + l15] = acc[t][r];
  }
}

// ---------------------------------------------------------------------------
// Launcher
// ---------------------------------------------------------------------------
extern "C" void kernel_launch(void* const* d_in, const int* in_sizes, int n_in,
                              void* d_out, int out_size, void* d_ws,
                              size_t ws_size, hipStream_t stream) {
  (void)in_sizes; (void)n_in; (void)out_size; (void)ws_size;

  const float* x    = (const float*)d_in[0];
  const int*   mask = (const int*)d_in[1];
  const float* Wq   = (const float*)d_in[2];
  const float* Wk   = (const float*)d_in[3];
  const float* Wv   = (const float*)d_in[4];
  const float* Wo   = (const float*)d_in[5];
  float* out = (float*)d_out;

  char* ws = (char*)d_ws;
  constexpr size_t XH_BYTES  = (size_t)B_ * S_ * D_ * 2;        // 8 MB
  constexpr size_t QKV_BYTES = (size_t)B_ * H_ * S_ * DH_ * 2;  // 8 MB each
  constexpr size_t WH_BYTES  = (size_t)H_ * D_ * DH_ * 2;       // 512 KB
  constexpr size_t WO_BYTES  = (size_t)D_ * D_ * 2;             // 512 KB

  size_t off = 0;
  __bf16* xh  = (__bf16*)(ws + off); off += XH_BYTES;
  __bf16* Qo  = (__bf16*)(ws + off); off += QKV_BYTES;
  __bf16* Ko  = (__bf16*)(ws + off); off += QKV_BYTES;
  __bf16* VoT = (__bf16*)(ws + off); off += QKV_BYTES;
  __bf16* ctx = (__bf16*)(ws + off); off += XH_BYTES;
  __bf16* WqT = (__bf16*)(ws + off); off += WH_BYTES;
  __bf16* WkT = (__bf16*)(ws + off); off += WH_BYTES;
  __bf16* WvT = (__bf16*)(ws + off); off += WH_BYTES;
  __bf16* WoT = (__bf16*)(ws + off); off += WO_BYTES;

  // 1) casts (+ weight transposes so GEMM B operands are k-major)
  {
    const int nx = B_ * S_ * D_;
    cast_f32_bf16<<<(nx + 255) / 256, 256, 0, stream>>>(x, xh, nx);
    const int nw = H_ * D_ * DH_;
    cast_transpose_bf16<<<(nw + 255) / 256, 256, 0, stream>>>(
        Wq, WqT, D_ * DH_, D_, DH_, nw);
    cast_transpose_bf16<<<(nw + 255) / 256, 256, 0, stream>>>(
        Wk, WkT, D_ * DH_, D_, DH_, nw);
    cast_transpose_bf16<<<(nw + 255) / 256, 256, 0, stream>>>(
        Wv, WvT, D_ * DH_, D_, DH_, nw);
    const int no = D_ * D_;
    cast_transpose_bf16<<<(no + 255) / 256, 256, 0, stream>>>(
        Wo, WoT, D_ * D_, D_, D_, no);
  }

  // 2) QKV projections (z selects Q/K/V)
  qkv_gemm<<<dim3(S_ / 64, B_ * H_, 3), 128, 0, stream>>>(
      xh, WqT, WkT, WvT, Qo, Ko, VoT);

  // 3) fused flash attention with key-padding mask (TDM-staged K/V)
  attn_kernel<<<dim3(S_ / 64, B_ * H_), 128, 0, stream>>>(
      Qo, Ko, VoT, mask, ctx);

  // 4) output projection -> f32
  out_gemm<<<dim3(B_ * S_ / 64, D_ / 64), 128, 0, stream>>>(ctx, WoT, out);
}
